// Diff_MultiHeadAttention_Optimized_7670811591187
// MI455X (gfx1250) — matrix operand order repaired and checked by reading, not compiled, not run on hardware
//
#include <hip/hip_runtime.h>
#include <hip/hip_bf16.h>
#include <stdint.h>

typedef __attribute__((ext_vector_type(16))) __bf16 v16bf;
typedef __attribute__((ext_vector_type(8)))  float  v8f;

#define N_NODES 20000
#define HDIM    256
#define HD      2048
#define BM      64
#define BN      128
#define BK      32
#define ASTR    40      // bf16 stride: 80B rows -> conflict-free, 16B aligned
#define BSTR    40
#define ABUF    (BM * ASTR)   // 2560 bf16
#define BBUF    (BN * BSTR)   // 5120 bf16
#define NTHREADS 256

// ---- fp32 -> {bf16 hi (RNE), bf16 lo (residual)} ---------------------------
static __device__ __forceinline__ void split_bits(float x, uint32_t& hb, uint32_t& lb) {
  uint32_t xb = __float_as_uint(x);
  hb = (xb + 0x7FFFu + ((xb >> 16) & 1u)) >> 16;
  float lo = x - __uint_as_float(hb << 16);
  lb = __float_as_uint(lo) >> 16;
}

union FC { uint4 q[2]; v16bf v; };

// ---- CDNA5 async global->LDS (ASYNCcnt), ISA 15.18.3 -----------------------
static __device__ __forceinline__ void async_ld128(uint32_t lds_off, const void* gaddr) {
  asm volatile("global_load_async_to_lds_b128 %0, %1, off"
               :: "v"(lds_off), "v"(gaddr) : "memory");
}
static __device__ __forceinline__ void wait_async_le6() {
  asm volatile("s_wait_asynccnt 0x6" ::: "memory");
}
static __device__ __forceinline__ void wait_async_le0() {
  asm volatile("s_wait_asynccnt 0x0" ::: "memory");
}

// ---- one BK=32 step: fragments are direct b128 loads, pre-packed -----------
static __device__ __forceinline__ void mma_step(const uint16_t* sAh, const uint16_t* sAl,
                                                const uint16_t* sBh, const uint16_t* sBl,
                                                int lane, int waveM, int waveN, v8f acc[2][2]) {
  const int r16 = lane & 15, h = lane >> 4;
  FC bh[2], bl[2];
#pragma unroll
  for (int sn = 0; sn < 2; ++sn) {
    const int cb = (waveN * 32 + sn * 16 + r16) * BSTR + 8 * h;    // K = {8h.., 16+8h..}
    bh[sn].q[0] = *(const uint4*)(sBh + cb);
    bh[sn].q[1] = *(const uint4*)(sBh + cb + 16);
    bl[sn].q[0] = *(const uint4*)(sBl + cb);
    bl[sn].q[1] = *(const uint4*)(sBl + cb + 16);
  }
#pragma unroll
  for (int sm = 0; sm < 2; ++sm) {
    const int rb = (waveM * 32 + sm * 16 + r16) * ASTR;            // K = {8h.., 16+8h..}
    FC ah, al;
    ah.q[0] = *(const uint4*)(sAh + rb + 8 * h);
    ah.q[1] = *(const uint4*)(sAh + rb + 16 + 8 * h);
    al.q[0] = *(const uint4*)(sAl + rb + 8 * h);
    al.q[1] = *(const uint4*)(sAl + rb + 16 + 8 * h);
#pragma unroll
    for (int sn = 0; sn < 2; ++sn) {
      acc[sm][sn] = __builtin_amdgcn_wmma_f32_16x16x32_bf16(false, ah.v, false, bh[sn].v, (short)0, acc[sm][sn], false, false);
      acc[sm][sn] = __builtin_amdgcn_wmma_f32_16x16x32_bf16(false, al.v, false, bh[sn].v, (short)0, acc[sm][sn], false, false);
      acc[sm][sn] = __builtin_amdgcn_wmma_f32_16x16x32_bf16(false, ah.v, false, bl[sn].v, (short)0, acc[sm][sn], false, false);
    }
  }
}

// ---- full K loop: double-buffered LDS, async staging -----------------------
// sA: [2 bufs][hi|lo][ABUF], sB: [2 bufs][hi|lo][BBUF]
static __device__ __forceinline__ void gemm_loop(
    const uint16_t* __restrict__ Agh, const uint16_t* __restrict__ Agl, size_t lda,
    const uint16_t* __restrict__ Bgh, const uint16_t* __restrict__ Bgl, size_t ldb,
    int K, int m0, int n0, int Mrows,
    uint16_t* sA, uint16_t* sB,
    v8f acc[2][2]) {
  const int tid = threadIdx.x;
  const int lane = tid & 31, waveM = (tid >> 5) >> 2, waveN = (tid >> 5) & 3;

  // A: one 8-bf16 slot per thread; clamp tail rows (their C rows are never stored)
  const int ai = tid >> 2, ajj = (tid & 3) << 3;
  int gm = m0 + ai; if (gm >= Mrows) gm = Mrows - 1;
  const size_t aoff = (size_t)gm * lda + ajj;
  const int a_l = ai * ASTR + ajj;
  // B: two 8-bf16 slots per thread
  const int c0 = tid >> 2, jj0 = (tid & 3) << 3;
  const int c1 = c0 + 64;
  const size_t boff0 = (size_t)(n0 + c0) * ldb + jj0;
  const size_t boff1 = (size_t)(n0 + c1) * ldb + jj0;
  const int b_l0 = c0 * BSTR + jj0, b_l1 = c1 * BSTR + jj0;

  uint32_t aL[2][2], bL0[2][2], bL1[2][2];
#pragma unroll
  for (int bf = 0; bf < 2; ++bf)
#pragma unroll
    for (int pl = 0; pl < 2; ++pl) {
      aL[bf][pl]  = (uint32_t)(uintptr_t)(sA + (bf * 2 + pl) * ABUF + a_l);
      bL0[bf][pl] = (uint32_t)(uintptr_t)(sB + (bf * 2 + pl) * BBUF + b_l0);
      bL1[bf][pl] = (uint32_t)(uintptr_t)(sB + (bf * 2 + pl) * BBUF + b_l1);
    }

  // prologue: stage k0 = 0 into buffer 0
  async_ld128(aL[0][0],  Agh + aoff);
  async_ld128(aL[0][1],  Agl + aoff);
  async_ld128(bL0[0][0], Bgh + boff0);
  async_ld128(bL0[0][1], Bgl + boff0);
  async_ld128(bL1[0][0], Bgh + boff1);
  async_ld128(bL1[0][1], Bgl + boff1);

  int p = 0;
  for (int k0 = 0; k0 < K; k0 += BK, p ^= 1) {
    const int kn = k0 + BK;
    if (kn < K) {                       // stage next into other buffer
      const int np = p ^ 1;
      async_ld128(aL[np][0],  Agh + aoff + kn);
      async_ld128(aL[np][1],  Agl + aoff + kn);
      async_ld128(bL0[np][0], Bgh + boff0 + kn);
      async_ld128(bL0[np][1], Bgl + boff0 + kn);
      async_ld128(bL1[np][0], Bgh + boff1 + kn);
      async_ld128(bL1[np][1], Bgl + boff1 + kn);
      wait_async_le6();                 // in-order: current buffer's 6 are done
    } else {
      wait_async_le0();
    }
    __syncthreads();                    // all waves' current-buffer data in LDS
    mma_step(sA + (p * 2) * ABUF, sA + (p * 2 + 1) * ABUF,
             sB + (p * 2) * BBUF, sB + (p * 2 + 1) * BBUF,
             lane, waveM, waveN, acc);
    __syncthreads();                    // done reading buf p before it is refilled
  }
}

// ---- epilogues -------------------------------------------------------------
static __device__ __forceinline__ void store_f32(float* C, size_t ldc, int M,
                                                 int m0, int n0, int lane, int waveM, int waveN,
                                                 const v8f acc[2][2], const float* bias) {
#pragma unroll
  for (int sm = 0; sm < 2; ++sm)
#pragma unroll
    for (int sn = 0; sn < 2; ++sn) {
      const int col = n0 + waveN * 32 + sn * 16 + (lane & 15);
      const float bv = bias ? bias[col] : 0.0f;
#pragma unroll
      for (int r = 0; r < 8; ++r) {
        const int row = m0 + waveM * 32 + sm * 16 + r + 8 * (lane >> 4);
        if (row < M) C[(size_t)row * ldc + col] = acc[sm][sn][r] + bv;
      }
    }
}

static __device__ __forceinline__ void store_rm_planes(uint16_t* Ch, uint16_t* Cl, size_t ldc, int M,
                                                       int m0, int n0, int lane, int waveM, int waveN,
                                                       const v8f acc[2][2], const float* bias) {
#pragma unroll
  for (int sm = 0; sm < 2; ++sm)
#pragma unroll
    for (int sn = 0; sn < 2; ++sn) {
      const int col = n0 + waveN * 32 + sn * 16 + (lane & 15);
      const float bv = bias ? bias[col] : 0.0f;
#pragma unroll
      for (int r = 0; r < 8; ++r) {
        const int row = m0 + waveM * 32 + sm * 16 + r + 8 * (lane >> 4);
        if (row < M) {
          uint32_t hb, lb; split_bits(acc[sm][sn][r] + bv, hb, lb);
          Ch[(size_t)row * ldc + col] = (uint16_t)hb;
          Cl[(size_t)row * ldc + col] = (uint16_t)lb;
        }
      }
    }
}

// transposed planes: per lane 8 consecutive rows -> one b128 store per plane
static __device__ __forceinline__ void store_tr_planes(uint16_t* Ch, uint16_t* Cl, size_t ldT, int Mrows,
                                                       int m0, int n0, int lane, int waveM, int waveN,
                                                       const v8f acc[2][2], const float* bias, float scale) {
#pragma unroll
  for (int sm = 0; sm < 2; ++sm)
#pragma unroll
    for (int sn = 0; sn < 2; ++sn) {
      const int col = n0 + waveN * 32 + sn * 16 + (lane & 15);
      const float bv = bias ? bias[col] : 0.0f;
      const int rowbase = m0 + waveM * 32 + sm * 16 + 8 * (lane >> 4);
      if (rowbase + 7 < Mrows) {                                 // 8-aligned, never straddles
        uint32_t hw[4], lw[4];
#pragma unroll
        for (int p = 0; p < 4; ++p) {
          uint32_t h0, l0, h1, l1;
          split_bits(acc[sm][sn][2 * p]     * scale + bv, h0, l0);
          split_bits(acc[sm][sn][2 * p + 1] * scale + bv, h1, l1);
          hw[p] = h0 | (h1 << 16);
          lw[p] = l0 | (l1 << 16);
        }
        *(uint4*)(Ch + (size_t)col * ldT + rowbase) = make_uint4(hw[0], hw[1], hw[2], hw[3]);
        *(uint4*)(Cl + (size_t)col * ldT + rowbase) = make_uint4(lw[0], lw[1], lw[2], lw[3]);
      }
    }
}

// ---- conversion kernels ----------------------------------------------------
__global__ void conv_rm(const float* __restrict__ src, uint16_t* __restrict__ dh,
                        uint16_t* __restrict__ dl, int n) {
  for (int i = blockIdx.x * blockDim.x + threadIdx.x; i < n; i += gridDim.x * blockDim.x) {
    uint32_t hb, lb; split_bits(src[i], hb, lb);
    dh[i] = (uint16_t)hb; dl[i] = (uint16_t)lb;
  }
}

__global__ void conv_tr(const float* __restrict__ src, uint16_t* __restrict__ dh,
                        uint16_t* __restrict__ dl, int R, int C) {
  const int total = R * C;
  for (int o = blockIdx.x * blockDim.x + threadIdx.x; o < total; o += gridDim.x * blockDim.x) {
    const int c = o / R, r = o - c * R;                          // dst[c][r] = src[r][c]
    uint32_t hb, lb; split_bits(src[(size_t)r * C + c], hb, lb);
    dh[o] = (uint16_t)hb; dl[o] = (uint16_t)lb;
  }
}

// ---- lambda scalars --------------------------------------------------------
__global__ void lambda_kernel(const float* __restrict__ lq1, const float* __restrict__ lk1,
                              const float* __restrict__ lq2, const float* __restrict__ lk2,
                              const int* __restrict__ layer, float* __restrict__ lam) {
  __shared__ float s1[NTHREADS], s2[NTHREADS];
  const int tid = threadIdx.x;
  s1[tid] = lq1[tid] * lk1[tid];
  s2[tid] = lq2[tid] * lk2[tid];
  __syncthreads();
  for (int off = NTHREADS / 2; off > 0; off >>= 1) {
    if (tid < off) { s1[tid] += s1[tid + off]; s2[tid] += s2[tid + off]; }
    __syncthreads();
  }
  if (tid == 0) {
    const float lam_init = 0.8f - 0.6f * __expf(-0.3f * (float)layer[0]);
    lam[0] = __expf(s1[0]) - __expf(s2[0]) + lam_init;
    lam[1] = 1.0f - lam_init;
  }
}

// ---- 5 projections ---------------------------------------------------------
__global__ void proj_kernel(
    const uint16_t* __restrict__ qh, const uint16_t* __restrict__ ql,
    const uint16_t* __restrict__ kh, const uint16_t* __restrict__ kl,
    const uint16_t* __restrict__ vh, const uint16_t* __restrict__ vl,
    const uint16_t* __restrict__ W1h, const uint16_t* __restrict__ W1l,
    const uint16_t* __restrict__ W2h, const uint16_t* __restrict__ W2l,
    const uint16_t* __restrict__ W3h, const uint16_t* __restrict__ W3l,
    const uint16_t* __restrict__ W4h, const uint16_t* __restrict__ W4l,
    const uint16_t* __restrict__ W5h, const uint16_t* __restrict__ W5l,
    const float* __restrict__ bq1, const float* __restrict__ bk1,
    const float* __restrict__ bq2, const float* __restrict__ bk2, const float* __restrict__ bv,
    uint16_t* __restrict__ q1h, uint16_t* __restrict__ q1l,
    uint16_t* __restrict__ k1h, uint16_t* __restrict__ k1l,   // transposed [HD][N]
    uint16_t* __restrict__ q2h, uint16_t* __restrict__ q2l,
    uint16_t* __restrict__ k2h, uint16_t* __restrict__ k2l,   // transposed
    uint16_t* __restrict__ vvh, uint16_t* __restrict__ vvl) { // transposed
  __shared__ __align__(16) uint16_t sA[2 * 2 * ABUF];
  __shared__ __align__(16) uint16_t sB[2 * 2 * BBUF];
  const int tid = threadIdx.x, lane = tid & 31;
  const int waveM = (tid >> 5) >> 2, waveN = (tid >> 5) & 3;
  const int m0 = blockIdx.x * BM, n0 = blockIdx.y * BN;
  const uint16_t *Ah, *Al, *Bh, *Bl; const float* bias;
  switch (blockIdx.z) {
    case 0:  Ah = qh; Al = ql; Bh = W1h; Bl = W1l; bias = bq1; break;
    case 1:  Ah = kh; Al = kl; Bh = W2h; Bl = W2l; bias = bk1; break;
    case 2:  Ah = qh; Al = ql; Bh = W3h; Bl = W3l; bias = bq2; break;
    case 3:  Ah = kh; Al = kl; Bh = W4h; Bl = W4l; bias = bk2; break;
    default: Ah = vh; Al = vl; Bh = W5h; Bl = W5l; bias = bv;  break;
  }
  v8f acc[2][2] = {};
  gemm_loop(Ah, Al, HDIM, Bh, Bl, HDIM, HDIM, m0, n0, N_NODES, sA, sB, acc);
  switch (blockIdx.z) {
    case 0:  store_rm_planes(q1h, q1l, HD, N_NODES, m0, n0, lane, waveM, waveN, acc, bias); break;
    case 1:  store_tr_planes(k1h, k1l, N_NODES, N_NODES, m0, n0, lane, waveM, waveN, acc, bias, 1.0f); break;
    case 2:  store_rm_planes(q2h, q2l, HD, N_NODES, m0, n0, lane, waveM, waveN, acc, bias); break;
    case 3:  store_tr_planes(k2h, k2l, N_NODES, N_NODES, m0, n0, lane, waveM, waveN, acc, bias, 1.0f); break;
    default: store_tr_planes(vvh, vvl, N_NODES, N_NODES, m0, n0, lane, waveM, waveN, acc, bias, 1.0f); break;
  }
}

// ---- kXv = kX^T @ vv, K = 20000; k2v pre-scaled by -lam --------------------
__global__ void kv_kernel(const uint16_t* __restrict__ k1h, const uint16_t* __restrict__ k1l,
                          const uint16_t* __restrict__ k2h, const uint16_t* __restrict__ k2l,
                          const uint16_t* __restrict__ vvh, const uint16_t* __restrict__ vvl,
                          const float* __restrict__ lam,
                          uint16_t* __restrict__ k1vh, uint16_t* __restrict__ k1vl,  // transposed [HD][HD]
                          uint16_t* __restrict__ k2vh, uint16_t* __restrict__ k2vl) {
  __shared__ __align__(16) uint16_t sA[2 * 2 * ABUF];
  __shared__ __align__(16) uint16_t sB[2 * 2 * BBUF];
  const int tid = threadIdx.x, lane = tid & 31;
  const int waveM = (tid >> 5) >> 2, waveN = (tid >> 5) & 3;
  const int m0 = blockIdx.x * BM, n0 = blockIdx.y * BN;
  const uint16_t* Ah = blockIdx.z ? k2h : k1h;
  const uint16_t* Al = blockIdx.z ? k2l : k1l;
  uint16_t* Ch = blockIdx.z ? k2vh : k1vh;
  uint16_t* Cl = blockIdx.z ? k2vl : k1vl;
  const float scale = blockIdx.z ? -lam[0] : 1.0f;
  v8f acc[2][2] = {};
  gemm_loop(Ah, Al, N_NODES, vvh, vvl, N_NODES, N_NODES, m0, n0, HD, sA, sB, acc);
  store_tr_planes(Ch, Cl, HD, HD, m0, n0, lane, waveM, waveN, acc, nullptr, scale);
}

// ---- x = q1@k1v + q2@(-lam*k2v) --------------------------------------------
__global__ void diff_kernel(const uint16_t* __restrict__ q1h, const uint16_t* __restrict__ q1l,
                            const uint16_t* __restrict__ q2h, const uint16_t* __restrict__ q2l,
                            const uint16_t* __restrict__ k1vh, const uint16_t* __restrict__ k1vl,
                            const uint16_t* __restrict__ k2vh, const uint16_t* __restrict__ k2vl,
                            float* __restrict__ x) {
  __shared__ __align__(16) uint16_t sA[2 * 2 * ABUF];
  __shared__ __align__(16) uint16_t sB[2 * 2 * BBUF];
  const int tid = threadIdx.x, lane = tid & 31;
  const int waveM = (tid >> 5) >> 2, waveN = (tid >> 5) & 3;
  const int m0 = blockIdx.x * BM, n0 = blockIdx.y * BN;
  v8f acc[2][2] = {};
  gemm_loop(q1h, q1l, HD, k1vh, k1vl, HD, HD, m0, n0, N_NODES, sA, sB, acc);
  gemm_loop(q2h, q2l, HD, k2vh, k2vl, HD, HD, m0, n0, N_NODES, sA, sB, acc);
  store_f32(x, HD, N_NODES, m0, n0, lane, waveM, waveN, acc, nullptr);
}

// ---- per-row LayerNorm + (1-lam_init) scale, emits split planes ------------
__global__ void ln_kernel(const float* __restrict__ x, const float* __restrict__ g,
                          const float* __restrict__ b, const float* __restrict__ lam,
                          uint16_t* __restrict__ xnh, uint16_t* __restrict__ xnl) {
  __shared__ float s1[NTHREADS], s2[NTHREADS];
  const int row = blockIdx.x, tid = threadIdx.x;
  float vals[HD / NTHREADS];
  float a = 0.0f, sq = 0.0f;
#pragma unroll
  for (int e = 0; e < HD / NTHREADS; ++e) {
    const float v = x[(size_t)row * HD + tid + NTHREADS * e];
    vals[e] = v; a += v; sq += v * v;
  }
  s1[tid] = a; s2[tid] = sq; __syncthreads();
  for (int off = NTHREADS / 2; off > 0; off >>= 1) {
    if (tid < off) { s1[tid] += s1[tid + off]; s2[tid] += s2[tid + off]; }
    __syncthreads();
  }
  const float mu  = s1[0] * (1.0f / HD);
  const float var = s2[0] * (1.0f / HD) - mu * mu;
  const float inv = rsqrtf(var + 1e-5f);
  const float sfac = lam[1];
#pragma unroll
  for (int e = 0; e < HD / NTHREADS; ++e) {
    const int col = tid + NTHREADS * e;
    const float o = ((vals[e] - mu) * inv * g[col] + b[col]) * sfac;
    uint32_t hb, lb; split_bits(o, hb, lb);
    xnh[(size_t)row * HD + col] = (uint16_t)hb;
    xnl[(size_t)row * HD + col] = (uint16_t)lb;
  }
}

// ---- out = xn @ Wo + bo ----------------------------------------------------
__global__ void out_kernel(const uint16_t* __restrict__ xnh, const uint16_t* __restrict__ xnl,
                           const uint16_t* __restrict__ woh, const uint16_t* __restrict__ wol,
                           const float* __restrict__ bo, float* __restrict__ out) {
  __shared__ __align__(16) uint16_t sA[2 * 2 * ABUF];
  __shared__ __align__(16) uint16_t sB[2 * 2 * BBUF];
  const int tid = threadIdx.x, lane = tid & 31;
  const int waveM = (tid >> 5) >> 2, waveN = (tid >> 5) & 3;
  const int m0 = blockIdx.x * BM, n0 = blockIdx.y * BN;
  v8f acc[2][2] = {};
  gemm_loop(xnh, xnl, HD, woh, wol, HD, HD, m0, n0, N_NODES, sA, sB, acc);
  store_f32(out, HDIM, N_NODES, m0, n0, lane, waveM, waveN, acc, bo);
}

extern "C" void kernel_launch(void* const* d_in, const int* in_sizes, int n_in,
                              void* d_out, int out_size, void* d_ws, size_t ws_size,
                              hipStream_t stream) {
  (void)in_sizes; (void)n_in; (void)out_size; (void)ws_size;
  const float* q   = (const float*)d_in[0];
  const float* k   = (const float*)d_in[1];
  const float* v   = (const float*)d_in[2];
  const int*   layer = (const int*)d_in[3];
  const float* Wq1 = (const float*)d_in[4];  const float* bq1 = (const float*)d_in[5];
  const float* Wk1 = (const float*)d_in[6];  const float* bk1 = (const float*)d_in[7];
  const float* Wq2 = (const float*)d_in[8];  const float* bq2 = (const float*)d_in[9];
  const float* Wk2 = (const float*)d_in[10]; const float* bk2 = (const float*)d_in[11];
  const float* Wv  = (const float*)d_in[12]; const float* bv  = (const float*)d_in[13];
  const float* ln_g = (const float*)d_in[14]; const float* ln_b = (const float*)d_in[15];
  const float* Wo  = (const float*)d_in[16]; const float* bo  = (const float*)d_in[17];
  const float* lq1 = (const float*)d_in[18]; const float* lk1 = (const float*)d_in[19];
  const float* lq2 = (const float*)d_in[20]; const float* lk2 = (const float*)d_in[21];
  float* out = (float*)d_out;

  // ---- workspace bump allocator (all plane sizes are multiples of 8 elems)
  const size_t NH  = (size_t)N_NODES * HDIM;  // 5.12M
  const size_t WH  = (size_t)HDIM * HD;       // 0.52M
  const size_t NHD = (size_t)N_NODES * HD;    // 40.96M
  const size_t HH  = (size_t)HD * HD;         // 4.19M
  uint8_t* wsb = (uint8_t*)d_ws;
  size_t off = 0;
  auto alloc16 = [&](size_t elems) { uint16_t* p = (uint16_t*)(wsb + off); off += elems * 2; return p; };
  auto allocf  = [&](size_t elems) { float*    p = (float*)(wsb + off);    off += elems * 4; return p; };

  uint16_t *qh = alloc16(NH), *ql = alloc16(NH);
  uint16_t *kh = alloc16(NH), *kl = alloc16(NH);
  uint16_t *vh = alloc16(NH), *vl = alloc16(NH);
  uint16_t *W1h = alloc16(WH), *W1l = alloc16(WH);
  uint16_t *W2h = alloc16(WH), *W2l = alloc16(WH);
  uint16_t *W3h = alloc16(WH), *W3l = alloc16(WH);
  uint16_t *W4h = alloc16(WH), *W4l = alloc16(WH);
  uint16_t *W5h = alloc16(WH), *W5l = alloc16(WH);
  uint16_t *woh = alloc16(WH), *wol = alloc16(WH);
  uint16_t *q1h = alloc16(NHD), *q1l = alloc16(NHD);
  uint16_t *q2h = alloc16(NHD), *q2l = alloc16(NHD);
  uint16_t *k1h = alloc16(NHD), *k1l = alloc16(NHD);   // transposed [HD][N]
  uint16_t *k2h = alloc16(NHD), *k2l = alloc16(NHD);   // transposed
  uint16_t *vvh = alloc16(NHD), *vvl = alloc16(NHD);   // transposed
  uint16_t *k1vh = alloc16(HH), *k1vl = alloc16(HH);   // transposed [HD][HD]
  uint16_t *k2vh = alloc16(HH), *k2vl = alloc16(HH);   // transposed
  float* xf = allocf(NHD);
  uint16_t *xnh = alloc16(NHD), *xnl = alloc16(NHD);
  float* lam = allocf(8);

  const int MT = (N_NODES + BM - 1) / BM;  // 313

  lambda_kernel<<<1, NTHREADS, 0, stream>>>(lq1, lk1, lq2, lk2, layer, lam);

  conv_rm<<<2048, NTHREADS, 0, stream>>>(q, qh, ql, (int)NH);
  conv_rm<<<2048, NTHREADS, 0, stream>>>(k, kh, kl, (int)NH);
  conv_rm<<<2048, NTHREADS, 0, stream>>>(v, vh, vl, (int)NH);
  conv_tr<<<512, NTHREADS, 0, stream>>>(Wq1, W1h, W1l, HDIM, HD);  // -> [HD][H]
  conv_tr<<<512, NTHREADS, 0, stream>>>(Wk1, W2h, W2l, HDIM, HD);
  conv_tr<<<512, NTHREADS, 0, stream>>>(Wq2, W3h, W3l, HDIM, HD);
  conv_tr<<<512, NTHREADS, 0, stream>>>(Wk2, W4h, W4l, HDIM, HD);
  conv_tr<<<512, NTHREADS, 0, stream>>>(Wv,  W5h, W5l, HDIM, HD);
  conv_tr<<<512, NTHREADS, 0, stream>>>(Wo,  woh, wol, HD, HDIM); // -> [H][HD]

  proj_kernel<<<dim3(MT, HD / BN, 5), NTHREADS, 0, stream>>>(
      qh, ql, kh, kl, vh, vl,
      W1h, W1l, W2h, W2l, W3h, W3l, W4h, W4l, W5h, W5l,
      bq1, bk1, bq2, bk2, bv,
      q1h, q1l, k1h, k1l, q2h, q2l, k2h, k2l, vvh, vvl);

  kv_kernel<<<dim3(HD / BM, HD / BN, 2), NTHREADS, 0, stream>>>(
      k1h, k1l, k2h, k2l, vvh, vvl, lam, k1vh, k1vl, k2vh, k2vl);

  diff_kernel<<<dim3(MT, HD / BN), NTHREADS, 0, stream>>>(
      q1h, q1l, q2h, q2l, k1vh, k1vl, k2vh, k2vl, xf);

  ln_kernel<<<N_NODES, NTHREADS, 0, stream>>>(xf, ln_g, ln_b, lam, xnh, xnl);

  out_kernel<<<dim3(MT, HDIM / BN), NTHREADS, 0, stream>>>(xnh, xnl, woh, wol, bo, out);
}
